// GATLayer_71236327572094
// MI455X (gfx1250) — compile-verified, matrix operand order checked by reading, not run on hardware
//
#include <hip/hip_runtime.h>

typedef __attribute__((ext_vector_type(16))) _Float16 v16h;
typedef __attribute__((ext_vector_type(8)))  _Float16 v8h;
typedef __attribute__((ext_vector_type(8)))  float    v8f;
typedef __attribute__((ext_vector_type(4)))  float    v4f;
typedef __attribute__((ext_vector_type(4)))  int      v4i;

#define GAT_ALPHA 0.2f
constexpr int N_NODES = 8192;
constexpr int F_IN    = 256;
constexpr int F_OUT   = 128;
constexpr int JSLICE  = 1024;   // j-range per wave (8 slices total)
constexpr int BROWS   = 128;    // rows per block (8 waves x 16)
constexpr int LDS_STRIDE = 40;  // 32 j + 8 pad (f16) -> conflict-free b128 reads

// CDNA5 async memory->LDS path (ASYNCcnt), guarded so we always compile.
#if __has_builtin(__builtin_amdgcn_global_load_async_to_lds_b128)
#define GAT_ASYNC_LDS 1
#define AS1 __attribute__((address_space(1)))
#define AS3 __attribute__((address_space(3)))
__device__ __forceinline__ void gat_wait_async() {
#if __has_builtin(__builtin_amdgcn_s_wait_asynccnt)
  __builtin_amdgcn_s_wait_asynccnt(0);
#else
  asm volatile("s_wait_asynccnt 0x0" ::: "memory");
#endif
}
#endif

// Monotone float->uint encoding so atomicMax(uint) == float max.
__device__ __forceinline__ unsigned enc_f32(float v) {
  unsigned u = __float_as_uint(v);
  return (u & 0x80000000u) ? ~u : (u | 0x80000000u);
}
__device__ __forceinline__ float dec_f32(unsigned k) {
  unsigned u = (k & 0x80000000u) ? (k & 0x7fffffffu) : ~k;
  return __uint_as_float(u);
}

// ---------------------------------------------------------------------------
// K0: zero the output accumulator + l[], transpose W (f32 256x128) -> WT (f16
// 128x256), init the f2-max key.
// ---------------------------------------------------------------------------
__global__ void gat_init(const float* __restrict__ W, _Float16* __restrict__ WT,
                         float* __restrict__ lsum, unsigned* __restrict__ key,
                         float* __restrict__ out) {
  const int i = blockIdx.x * 256 + threadIdx.x;          // 0 .. 1048575
  out[i] = 0.0f;                                          // numerator accumulator
  if (i < F_IN * F_OUT) {                                 // W transpose, f32->f16
    const int k = i >> 7;          // 0..255
    const int n = i & 127;         // 0..127
    WT[n * F_IN + k] = (_Float16)W[i];
  }
  if (i < N_NODES) lsum[i] = 0.0f;
  if (i == 0) *key = 0u;                                  // < any encoded float
}

// ---------------------------------------------------------------------------
// K1: Wh = h @ W via v_wmma_f32_16x16x32_f16, stored transposed as f16.
// ---------------------------------------------------------------------------
__global__ void wh_gemm(const float* __restrict__ h, const _Float16* __restrict__ WT,
                        _Float16* __restrict__ WhT) {
  const int lane = threadIdx.x & 31;
  const int wv   = threadIdx.x >> 5;      // N-tile 0..7
  const int r    = lane & 15;
  const int hl   = lane >> 4;
  const int m_base = blockIdx.x * 16;
  const int n_base = wv * 16;

  const float*    hrow = h  + (size_t)(m_base + r) * F_IN;
  const _Float16* wrow = WT + (size_t)(n_base + r) * F_IN;

  v8f c = {};
  for (int k = 0; k < F_IN; k += 32) {
    v4f a0 = *(const v4f*)(hrow + k + hl * 8);
    v4f a1 = *(const v4f*)(hrow + k + hl * 8 + 4);
    v4f a2 = *(const v4f*)(hrow + k + 16 + hl * 8);
    v4f a3 = *(const v4f*)(hrow + k + 16 + hl * 8 + 4);
    v16h av;
#pragma unroll
    for (int t = 0; t < 4; ++t) {
      av[t]      = (_Float16)a0[t];
      av[4 + t]  = (_Float16)a1[t];
      av[8 + t]  = (_Float16)a2[t];
      av[12 + t] = (_Float16)a3[t];
    }
    v8h b0 = *(const v8h*)(wrow + k + hl * 8);
    v8h b1 = *(const v8h*)(wrow + k + 16 + hl * 8);
    v16h bv;
#pragma unroll
    for (int t = 0; t < 8; ++t) { bv[t] = b0[t]; bv[8 + t] = b1[t]; }

    c = __builtin_amdgcn_wmma_f32_16x16x32_f16(false, av, false, bv,
                                               (short)0, c, false, false);
  }
  v8h oh;
#pragma unroll
  for (int g = 0; g < 8; ++g) oh[g] = (_Float16)c[g];
  *(v8h*)(WhT + (size_t)(n_base + r) * N_NODES + m_base + hl * 8) = oh;
}

// ---------------------------------------------------------------------------
// K2: f1 = Wh@a1, f2 = Wh@a2 (from WhT f16), plus global max(f2) via uint key.
// ---------------------------------------------------------------------------
__global__ void f1f2_kernel(const _Float16* __restrict__ WhT,
                            const float* __restrict__ a,
                            float* __restrict__ f1, float* __restrict__ f2,
                            unsigned* __restrict__ key) {
  const int m = blockIdx.x * 256 + threadIdx.x;
  float s1 = 0.0f, s2 = 0.0f;
#pragma unroll 4
  for (int f = 0; f < F_OUT; ++f) {
    const float w = (float)WhT[(size_t)f * N_NODES + m];
    s1 += w * a[f];
    s2 += w * a[F_OUT + f];
  }
  f1[m] = s1;
  f2[m] = s2;
  atomicMax(key, enc_f32(s2));
}

// ---------------------------------------------------------------------------
// K3: fused masked-softmax-attention * Wh with static row max.
// Double-buffered WhT slab staging via async memory->LDS DMA (ASYNCcnt) when
// available; one barrier per 32-j chunk; DMA of chunk i+1 overlaps compute i.
// ---------------------------------------------------------------------------
__global__ void gat_attn(const int* __restrict__ adj,
                         const _Float16* __restrict__ WhT,
                         const float* __restrict__ f1,
                         const float* __restrict__ f2,
                         const unsigned* __restrict__ key,
                         float* __restrict__ outnum,
                         float* __restrict__ lsum) {
  __shared__ _Float16 Bs[2][F_OUT * LDS_STRIDE];   // 2 x 10 KB

  const int tid  = threadIdx.x;
  const int lane = tid & 31;
  const int wv   = tid >> 5;
  const int r    = lane & 15;       // A-operand row within tile
  const int hl   = lane >> 4;       // lane half -> K split
  const int row  = blockIdx.x * BROWS + wv * 16 + r;
  const int jbeg = blockIdx.y * JSLICE;
  const int jend = jbeg + JSLICE;

  const float f2max = dec_f32(*key);
  const float f1r   = f1[row];
  const float xm    = f1r + f2max;
  const float m_i   = xm > 0.0f ? xm : GAT_ALPHA * xm;   // upper bound on row max

  const int* arow = adj + (size_t)row * N_NODES;

  float rl = 0.0f;
  v8f acc[8];
#pragma unroll
  for (int nt = 0; nt < 8; ++nt) acc[nt] = (v8f){};

  // staging assignment: thread t -> feature row t>>1, 16-f16 segment t&1
  const int sn = tid >> 1;
  const int sg = (tid & 1) * 16;

  auto stage = [&](int buf, int jc) {
    const _Float16* src = WhT + (size_t)sn * N_NODES + jc + sg;
    _Float16*       dst = &Bs[buf][sn * LDS_STRIDE + sg];
#ifdef GAT_ASYNC_LDS
    __builtin_amdgcn_global_load_async_to_lds_b128(
        (AS1 v4i*)src,       (AS3 v4i*)dst,       0, 0);
    __builtin_amdgcn_global_load_async_to_lds_b128(
        (AS1 v4i*)(src + 8), (AS3 v4i*)(dst + 8), 0, 0);
#else
    *(v8h*)dst       = *(const v8h*)src;
    *(v8h*)(dst + 8) = *(const v8h*)(src + 8);
#endif
  };

  stage(0, jbeg);                   // prologue: slab 0 in flight
  int buf = 0;

  for (int jc = jbeg; jc < jend; jc += 32, buf ^= 1) {
#ifdef GAT_ASYNC_LDS
    gat_wait_async();               // own slab-i DMA complete
#endif
    __syncthreads();                // all waves: slab i landed, slab i-1 consumed
    if (jc + 32 < jend) {
      stage(buf ^ 1, jc + 32);      // DMA of slab i+1 overlaps compute of slab i
      __builtin_prefetch(arow + jc + 32 + hl * 8, 0, 3);   // adj stream ahead
    }

    // ---- masked scores -> P (A layout: elem i -> K=(i>>3)*16+hl*8+(i&7))
    v4i mk0 = *(const v4i*)(arow + jc + hl * 8);
    v4i mk1 = *(const v4i*)(arow + jc + hl * 8 + 4);
    v4i mk2 = *(const v4i*)(arow + jc + 16 + hl * 8);
    v4i mk3 = *(const v4i*)(arow + jc + 16 + hl * 8 + 4);
    v4f q0  = *(const v4f*)(f2 + jc + hl * 8);
    v4f q1  = *(const v4f*)(f2 + jc + hl * 8 + 4);
    v4f q2  = *(const v4f*)(f2 + jc + 16 + hl * 8);
    v4f q3  = *(const v4f*)(f2 + jc + 16 + hl * 8 + 4);

    v16h pv;
#pragma unroll
    for (int t = 0; t < 4; ++t) {
      float x, e, p;
      x = f1r + q0[t]; e = x > 0.0f ? x : GAT_ALPHA * x;
      p = (mk0[t] > 0) ? __expf(e - m_i) : 0.0f; rl += p; pv[t] = (_Float16)p;
      x = f1r + q1[t]; e = x > 0.0f ? x : GAT_ALPHA * x;
      p = (mk1[t] > 0) ? __expf(e - m_i) : 0.0f; rl += p; pv[4 + t] = (_Float16)p;
      x = f1r + q2[t]; e = x > 0.0f ? x : GAT_ALPHA * x;
      p = (mk2[t] > 0) ? __expf(e - m_i) : 0.0f; rl += p; pv[8 + t] = (_Float16)p;
      x = f1r + q3[t]; e = x > 0.0f ? x : GAT_ALPHA * x;
      p = (mk3[t] > 0) ? __expf(e - m_i) : 0.0f; rl += p; pv[12 + t] = (_Float16)p;
    }

    // ---- P @ Wh[jc:jc+32, :]  (8 N-tiles), B from LDS
#pragma unroll
    for (int nt = 0; nt < 8; ++nt) {
      const _Float16* bp = &Bs[buf][(nt * 16 + r) * LDS_STRIDE + hl * 8];
      v8h b0 = *(const v8h*)bp;
      v8h b1 = *(const v8h*)(bp + 16);
      v16h bv;
#pragma unroll
      for (int t = 0; t < 8; ++t) { bv[t] = b0[t]; bv[8 + t] = b1[t]; }
      acc[nt] = __builtin_amdgcn_wmma_f32_16x16x32_f16(false, pv, false, bv,
                                                       (short)0, acc[nt], false, false);
    }
  }

  // denominator: lanes L and L^16 cover the same row, disjoint j's
  rl += __shfl_xor(rl, 16, 32);
  if (hl == 0) atomicAdd(&lsum[row], rl);

  // numerator: C/D layout lane -> N = r, M = g + 8*hl
  const int orow0 = blockIdx.x * BROWS + wv * 16 + hl * 8;
#pragma unroll
  for (int nt = 0; nt < 8; ++nt) {
#pragma unroll
    for (int g = 0; g < 8; ++g) {
      atomicAdd(&outnum[(size_t)(orow0 + g) * F_OUT + nt * 16 + r], acc[nt][g]);
    }
  }
}

// ---------------------------------------------------------------------------
// K4: h_prime = numerator / l
// ---------------------------------------------------------------------------
__global__ void gat_finalize(float* __restrict__ out, const float* __restrict__ lsum) {
  const int i = blockIdx.x * 256 + threadIdx.x;   // 0 .. 1048575
  out[i] = out[i] / lsum[i >> 7];
}

// ---------------------------------------------------------------------------
extern "C" void kernel_launch(void* const* d_in, const int* in_sizes, int n_in,
                              void* d_out, int out_size, void* d_ws, size_t ws_size,
                              hipStream_t stream) {
  (void)in_sizes; (void)n_in; (void)out_size; (void)ws_size;
  const float* h   = (const float*)d_in[0];
  const int*   adj = (const int*)d_in[1];
  const float* W   = (const float*)d_in[2];
  const float* a   = (const float*)d_in[3];
  float* out = (float*)d_out;

  char* ws = (char*)d_ws;
  size_t off = 0;
  _Float16* WhT = (_Float16*)(ws + off); off += (size_t)F_OUT * N_NODES * sizeof(_Float16); // 2 MB
  _Float16* WT  = (_Float16*)(ws + off); off += (size_t)F_OUT * F_IN * sizeof(_Float16);    // 64 KB
  float* f1   = (float*)(ws + off); off += (size_t)N_NODES * sizeof(float);
  float* f2   = (float*)(ws + off); off += (size_t)N_NODES * sizeof(float);
  float* lsum = (float*)(ws + off); off += (size_t)N_NODES * sizeof(float);
  unsigned* key = (unsigned*)(ws + off); off += 64;

  gat_init    <<<4096, 256, 0, stream>>>(W, WT, lsum, key, out);
  wh_gemm     <<<N_NODES / 16, 256, 0, stream>>>(h, WT, WhT);
  f1f2_kernel <<<N_NODES / 256, 256, 0, stream>>>(WhT, a, f1, f2, key);
  gat_attn    <<<dim3(N_NODES / BROWS, N_NODES / JSLICE), 256, 0, stream>>>(
                  adj, WhT, f1, f2, key, out, lsum);
  gat_finalize<<<(N_NODES * F_OUT) / 256, 256, 0, stream>>>(out, lsum);
}
